// FeaturePropagation_22522808500493
// MI455X (gfx1250) — matrix operand order repaired and checked by reading, not compiled, not run on hardware
//
#include <hip/hip_runtime.h>
#include <hip/hip_bf16.h>

typedef __bf16 bf16_t;
typedef __attribute__((ext_vector_type(16))) __bf16 v16bf;
typedef __attribute__((ext_vector_type(8)))  float  v8f;

// Problem constants (from reference setup_inputs)
#define BB   16
#define NN   4096
#define MM   1024
#define D1   128
#define D2   256
#define CIN  384
#define COUT 256
#define RTOT (BB * NN)          // 65536 rows
#define BN_EPS 1e-5f

// gfx1250 async global->LDS staging (ASYNCcnt path). Set to 0 to fall back to
// register staging if the assembler rejects the mnemonic.
#define USE_ASYNC_COPY 1

// ---------------------------------------------------------------------------
// Kernel 1: 3-NN + inverse-distance weights.
// ---------------------------------------------------------------------------
__global__ void fp_knn_kernel(const float* __restrict__ coords_1,
                              const float* __restrict__ coords_2,
                              int* __restrict__ idx_out,
                              float* __restrict__ w_out) {
    __shared__ __attribute__((aligned(16))) float sc[MM * 3];
    const int b  = blockIdx.x >> 4;            // 16 blocks per batch
    const int n0 = (blockIdx.x & 15) * 256;
    const int tid = threadIdx.x;

    const float* c2 = coords_2 + (size_t)b * MM * 3;
    for (int i = tid; i < MM * 3; i += 256) sc[i] = c2[i];
    __syncthreads();

    const int n = n0 + tid;
    const size_t p = (size_t)b * NN + n;
    const float qx = coords_1[p * 3 + 0];
    const float qy = coords_1[p * 3 + 1];
    const float qz = coords_1[p * 3 + 2];

    float d0 = 3.4e38f, dd1 = 3.4e38f, d2 = 3.4e38f;
    int   i0 = 0,       i1  = 0,       i2 = 0;
    for (int m = 0; m < MM; ++m) {
        const float dx = qx - sc[m * 3 + 0];
        const float dy = qy - sc[m * 3 + 1];
        const float dz = qz - sc[m * 3 + 2];
        const float d  = dx * dx + dy * dy + dz * dz;
        if (d < d0)       { d2 = dd1; i2 = i1; dd1 = d0; i1 = i0; d0 = d;  i0 = m; }
        else if (d < dd1) { d2 = dd1; i2 = i1; dd1 = d;  i1 = m; }
        else if (d < d2)  { d2 = d;  i2 = m; }
    }
    float w0 = 1.0f / (d0 + 1e-9f);
    float w1 = 1.0f / (dd1 + 1e-9f);
    float w2 = 1.0f / (d2 + 1e-9f);
    const float inv = 1.0f / (w0 + w1 + w2);
    w0 *= inv; w1 *= inv; w2 *= inv;
    idx_out[p * 3 + 0] = i0; idx_out[p * 3 + 1] = i1; idx_out[p * 3 + 2] = i2;
    w_out[p * 3 + 0] = w0;  w_out[p * 3 + 1] = w1;  w_out[p * 3 + 2] = w2;
}

// ---------------------------------------------------------------------------
// Kernel 2: build concatenated bf16 input X = [features_1 | interp] (R x 384)
// ---------------------------------------------------------------------------
__global__ void fp_build_x_kernel(const float* __restrict__ features_1,
                                  const float* __restrict__ features_2,
                                  const int* __restrict__ idx_in,
                                  const float* __restrict__ w_in,
                                  bf16_t* __restrict__ X) {
    const int row = blockIdx.x;
    const int b   = row >> 12;                 // row / NN
    const int j0 = idx_in[row * 3 + 0];
    const int j1 = idx_in[row * 3 + 1];
    const int j2 = idx_in[row * 3 + 2];
    const float w0 = w_in[row * 3 + 0];
    const float w1 = w_in[row * 3 + 1];
    const float w2 = w_in[row * 3 + 2];
    const float* f2b = features_2 + (size_t)b * MM * D2;

    for (int col = threadIdx.x; col < CIN; col += 128) {
        float v;
        if (col < D1) {
            v = features_1[(size_t)row * D1 + col];
        } else {
            const int c = col - D1;
            v = w0 * f2b[(size_t)j0 * D2 + c]
              + w1 * f2b[(size_t)j1 * D2 + c]
              + w2 * f2b[(size_t)j2 * D2 + c];
        }
        X[(size_t)row * CIN + col] = (bf16_t)v;
    }
}

// ---------------------------------------------------------------------------
// Kernel 3: W (K x Nc, fp32, row-major) -> Wt (Nc x K, bf16) transpose+convert
// ---------------------------------------------------------------------------
__global__ void fp_convert_w_kernel(const float* __restrict__ W,
                                    bf16_t* __restrict__ Wt,
                                    int K, int Nc) {
    const int e = blockIdx.x * 256 + threadIdx.x;
    if (e >= K * Nc) return;
    const int k = e / Nc;
    const int n = e - k * Nc;
    Wt[(size_t)n * K + k] = (bf16_t)W[e];
}

// ---------------------------------------------------------------------------
// Kernel 4: GEMM  Y(R x 256) = X(R x K, bf16) * Wt^T + bias, fp32 accumulate.
// 128x64 block tile, 256 threads = 8 waves (4 row x 2 col). Each wave owns a
// 32x32 sub-tile: 2 A-frags x 2 B-frags -> 4 v_wmma_f32_16x16x32_bf16 / K-step.
// Tiles staged with GLOBAL_LOAD_ASYNC_TO_LDS_B128 (ASYNCcnt), double-buffered.
// ---------------------------------------------------------------------------
__device__ inline v16bf frag_from_lds_32B(const bf16_t* p0, const bf16_t* p1) {
    union { uint4 u[2]; v16bf v; } r;
    r.u[0] = *reinterpret_cast<const uint4*>(p0);
    r.u[1] = *reinterpret_cast<const uint4*>(p1);
    return r.v;
}

#if USE_ASYNC_COPY
__device__ inline void async_b128_to_lds(const bf16_t* gsrc, unsigned lds_addr) {
    asm volatile("global_load_async_to_lds_b128 %0, %1, off"
                 :: "v"(lds_addr), "v"(gsrc) : "memory");
}
__device__ inline void wait_asynccnt0() {
    asm volatile("s_wait_asynccnt 0x0" ::: "memory");
}
#endif

__global__ void __launch_bounds__(256)
fp_gemm_kernel(const bf16_t* __restrict__ A,   // R x K (row-major bf16)
               const bf16_t* __restrict__ Wt,  // 256 x K (row-major bf16)
               const float* __restrict__ bias, // 256
               float* __restrict__ Y,          // R x 256 fp32
               int K) {
    __shared__ __attribute__((aligned(16))) bf16_t Alds[2][128 * 32]; // 16 KB
    __shared__ __attribute__((aligned(16))) bf16_t Blds[2][64 * 32];  //  8 KB

    const int tid     = threadIdx.x;
    const int lane    = tid & 31;
    const int wave    = tid >> 5;       // 0..7
    const int wm      = wave & 3;       // wave row group (32 rows each)
    const int wn      = wave >> 2;      // wave col group (32 cols each)
    const int colBase = blockIdx.x * 64;
    const int rowBase = blockIdx.y * 128;

    const int tr    = tid >> 2;         // 0..63
    const int chunk = tid & 3;          // 0..3  (8 bf16 per chunk)

    const int nK = K >> 5;              // K / 32

    // Per-thread global sources for tile staging (A: two rows, B: one col-row)
    const bf16_t* aG0 = A  + (size_t)(rowBase + tr)      * K + chunk * 8;
    const bf16_t* aG1 = A  + (size_t)(rowBase + 64 + tr) * K + chunk * 8;
    const bf16_t* bG  = Wt + (size_t)(colBase + tr)      * K + chunk * 8;

#if USE_ASYNC_COPY
    // LDS byte addresses (generic shared pointer low 32 bits == LDS offset)
    const unsigned aL = (unsigned)(uintptr_t)&Alds[0][tr * 32 + chunk * 8];
    const unsigned bL = (unsigned)(uintptr_t)&Blds[0][tr * 32 + chunk * 8];
    const unsigned aBufStride = 128 * 32 * 2;   // bytes per A buffer
    const unsigned bBufStride = 64 * 32 * 2;    // bytes per B buffer

    // issue tile kk into buffer `buf`
    auto issue_tile = [&](int buf, int kk) {
        const unsigned ao = aL + (unsigned)buf * aBufStride;
        const unsigned bo = bL + (unsigned)buf * bBufStride;
        async_b128_to_lds(aG0 + (size_t)kk * 32, ao);
        async_b128_to_lds(aG1 + (size_t)kk * 32, ao + 64 * 32 * 2);
        async_b128_to_lds(bG  + (size_t)kk * 32, bo);
    };

    issue_tile(0, 0);
    wait_asynccnt0();
    __syncthreads();
#else
    {
        uint4 a0 = *reinterpret_cast<const uint4*>(aG0);
        uint4 a1 = *reinterpret_cast<const uint4*>(aG1);
        uint4 bv = *reinterpret_cast<const uint4*>(bG);
        *reinterpret_cast<uint4*>(&Alds[0][tr * 32 + chunk * 8]) = a0;
        *reinterpret_cast<uint4*>(&Alds[0][(64 + tr) * 32 + chunk * 8]) = a1;
        *reinterpret_cast<uint4*>(&Blds[0][tr * 32 + chunk * 8]) = bv;
    }
    __syncthreads();
#endif

    v8f acc00 = {0.f, 0.f, 0.f, 0.f, 0.f, 0.f, 0.f, 0.f};
    v8f acc01 = acc00, acc10 = acc00, acc11 = acc00;

    // A-fragment addressing (16-bit A 16x32 layout):
    //   lane<16 : M=lane,    K in {kb..kb+7} U {kb+16..kb+23}, kb=0
    //   lane>=16: M=lane-16, same with kb=8
    const int ar0 = wm * 32 + (lane & 15);
    const int ar1 = ar0 + 16;
    const int kb  = (lane >> 4) * 8;
    // B-fragment addressing (32x16, column-major-in-LDS; same K split as A)
    const int kh  = (lane >> 4) * 8;
    const int bc0 = (wn * 32 + 0)  + (lane & 15);
    const int bc1 = (wn * 32 + 16) + (lane & 15);

    int cur = 0;
    for (int kk = 0; kk < nK; ++kk) {
        const bool more = (kk + 1) < nK;

#if USE_ASYNC_COPY
        if (more) issue_tile(cur ^ 1, kk + 1);
#else
        uint4 a0s, a1s, bvs;
        if (more) {
            a0s = *reinterpret_cast<const uint4*>(aG0 + (size_t)(kk + 1) * 32);
            a1s = *reinterpret_cast<const uint4*>(aG1 + (size_t)(kk + 1) * 32);
            bvs = *reinterpret_cast<const uint4*>(bG  + (size_t)(kk + 1) * 32);
            if ((kk + 2) < nK) {
                __builtin_prefetch(aG0 + (size_t)(kk + 2) * 32, 0, 1);
                __builtin_prefetch(bG  + (size_t)(kk + 2) * 32, 0, 1);
            }
        }
#endif

        const bf16_t* pa0 = &Alds[cur][ar0 * 32 + kb];
        const bf16_t* pa1 = &Alds[cur][ar1 * 32 + kb];
        v16bf af0 = frag_from_lds_32B(pa0, pa0 + 16);
        v16bf af1 = frag_from_lds_32B(pa1, pa1 + 16);
        const bf16_t* pb0 = &Blds[cur][bc0 * 32 + kh];
        const bf16_t* pb1 = &Blds[cur][bc1 * 32 + kh];
        v16bf bf0 = frag_from_lds_32B(pb0, pb0 + 16);
        v16bf bf1 = frag_from_lds_32B(pb1, pb1 + 16);

        acc00 = __builtin_amdgcn_wmma_f32_16x16x32_bf16(
            false, af0, false, bf0, (short)0, acc00, false, false);
        acc01 = __builtin_amdgcn_wmma_f32_16x16x32_bf16(
            false, af0, false, bf1, (short)0, acc01, false, false);
        acc10 = __builtin_amdgcn_wmma_f32_16x16x32_bf16(
            false, af1, false, bf0, (short)0, acc10, false, false);
        acc11 = __builtin_amdgcn_wmma_f32_16x16x32_bf16(
            false, af1, false, bf1, (short)0, acc11, false, false);

        if (more) {
#if USE_ASYNC_COPY
            wait_asynccnt0();
            __syncthreads();
#else
            __syncthreads();
            const int nxt = cur ^ 1;
            *reinterpret_cast<uint4*>(&Alds[nxt][tr * 32 + chunk * 8]) = a0s;
            *reinterpret_cast<uint4*>(&Alds[nxt][(64 + tr) * 32 + chunk * 8]) = a1s;
            *reinterpret_cast<uint4*>(&Blds[nxt][tr * 32 + chunk * 8]) = bvs;
            __syncthreads();
#endif
            cur ^= 1;
        }
    }

    // Epilogue: C layout — VGPR i, lane l: M = i + 8*(l>>4), N = l&15
    const int m_off = (lane >> 4) * 8;
    const int n_in  = lane & 15;
    const int gc0   = colBase + wn * 32 + n_in;
    const int gc1   = gc0 + 16;
    const float bv0 = bias[gc0];
    const float bv1 = bias[gc1];
    #pragma unroll
    for (int i = 0; i < 8; ++i) {
        const int grow0 = rowBase + wm * 32 + m_off + i;
        const int grow1 = grow0 + 16;
        Y[(size_t)grow0 * COUT + gc0] = acc00[i] + bv0;
        Y[(size_t)grow0 * COUT + gc1] = acc01[i] + bv1;
        Y[(size_t)grow1 * COUT + gc0] = acc10[i] + bv0;
        Y[(size_t)grow1 * COUT + gc1] = acc11[i] + bv1;
    }
}

// ---------------------------------------------------------------------------
// Kernel 5: per-channel sum / sumsq partial reduction (coalesced), atomics.
// ---------------------------------------------------------------------------
__global__ void fp_bn_stats_kernel(const float* __restrict__ Y,
                                   float* __restrict__ stats) {
    const int c = threadIdx.x;
    const size_t base = (size_t)blockIdx.x * 256 * COUT;
    float s = 0.f, q = 0.f;
    for (int r = 0; r < 256; ++r) {
        const float v = Y[base + (size_t)r * COUT + c];
        s += v;
        q += v * v;
    }
    atomicAdd(&stats[c], s);
    atomicAdd(&stats[COUT + c], q);
}

// ---------------------------------------------------------------------------
// Kernel 6: BN (training stats) + ReLU; emit bf16 (next GEMM) or fp32 (output)
// ---------------------------------------------------------------------------
template <bool OUT_BF16>
__global__ void fp_bn_apply_kernel(const float* __restrict__ Y,
                                   const float* __restrict__ stats,
                                   const float* __restrict__ gamma,
                                   const float* __restrict__ beta,
                                   bf16_t* __restrict__ out_bf,
                                   float* __restrict__ out_f) {
    const size_t e = (size_t)blockIdx.x * 256 + threadIdx.x;
    const int col = (int)(e & (COUT - 1));
    const float invR = 1.0f / (float)RTOT;
    const float mean = stats[col] * invR;
    const float var  = stats[COUT + col] * invR - mean * mean;
    const float sc   = gamma[col] * rsqrtf(var + BN_EPS);
    float y = sc * (Y[e] - mean) + beta[col];
    y = fmaxf(y, 0.0f);
    if (OUT_BF16) out_bf[e] = (bf16_t)y;
    else          out_f[e]  = y;
}

// ---------------------------------------------------------------------------
extern "C" void kernel_launch(void* const* d_in, const int* in_sizes, int n_in,
                              void* d_out, int out_size, void* d_ws, size_t ws_size,
                              hipStream_t stream) {
    const float* coords_1   = (const float*)d_in[0];
    const float* coords_2   = (const float*)d_in[1];
    const float* features_1 = (const float*)d_in[2];
    const float* features_2 = (const float*)d_in[3];
    const float* W1     = (const float*)d_in[4];
    const float* b1     = (const float*)d_in[5];
    const float* gamma1 = (const float*)d_in[6];
    const float* beta1  = (const float*)d_in[7];
    const float* W2     = (const float*)d_in[8];
    const float* b2     = (const float*)d_in[9];
    const float* gamma2 = (const float*)d_in[10];
    const float* beta2  = (const float*)d_in[11];
    float* out = (float*)d_out;

    // Workspace layout
    char* ws = (char*)d_ws;
    float*  stats = (float*)ws;                                   // 4096 B
    int*    idxb  = (int*)  (ws + 4096);                          // 65536*3*4
    float*  wgt   = (float*)(ws + 4096 + 786432);                 // 65536*3*4
    bf16_t* W1t   = (bf16_t*)(ws + 4096 + 2 * 786432);            // 384*256*2
    bf16_t* W2t   = (bf16_t*)(ws + 1773568);                      // 256*256*2
    bf16_t* X     = (bf16_t*)(ws + 1904640);                      // 65536*384*2 (reused as X2)
    float*  Y     = (float*) (ws + 52236288);                     // 65536*256*4

    float* stats1 = stats;
    float* stats2 = stats + 512;

    hipMemsetAsync(stats, 0, 4096, stream);

    // weights -> transposed bf16
    fp_convert_w_kernel<<<(CIN * COUT) / 256, 256, 0, stream>>>(W1, W1t, CIN, COUT);
    fp_convert_w_kernel<<<(COUT * COUT) / 256, 256, 0, stream>>>(W2, W2t, COUT, COUT);

    // 3-NN interpolation weights
    fp_knn_kernel<<<RTOT / 256, 256, 0, stream>>>(coords_1, coords_2, idxb, wgt);

    // concatenated bf16 input
    fp_build_x_kernel<<<RTOT, 128, 0, stream>>>(features_1, features_2, idxb, wgt, X);

    // Layer 1: GEMM + bias -> Y ; stats ; BN+ReLU -> bf16 X2 (aliases X)
    fp_gemm_kernel<<<dim3(COUT / 64, RTOT / 128), 256, 0, stream>>>(X, W1t, b1, Y, CIN);
    fp_bn_stats_kernel<<<RTOT / 256, 256, 0, stream>>>(Y, stats1);
    bf16_t* X2 = X;
    fp_bn_apply_kernel<true><<<(RTOT * COUT) / 256, 256, 0, stream>>>(
        Y, stats1, gamma1, beta1, X2, nullptr);

    // Layer 2: GEMM + bias -> Y ; stats ; BN+ReLU -> fp32 output
    fp_gemm_kernel<<<dim3(COUT / 64, RTOT / 128), 256, 0, stream>>>(X2, W2t, b2, Y, COUT);
    fp_bn_stats_kernel<<<RTOT / 256, 256, 0, stream>>>(Y, stats2);
    fp_bn_apply_kernel<false><<<(RTOT * COUT) / 256, 256, 0, stream>>>(
        Y, stats2, gamma2, beta2, nullptr, out);
}
